// HandwritingSynthesisNetwork_55241869361602
// MI455X (gfx1250) — compile-verified, matrix operand order checked by reading, not run on hardware
//
#include <hip/hip_runtime.h>
#include <hip/hip_bf16.h>

// ---------------- problem constants ----------------
#define NBLK 25      // persistent workgroups (each owns 16 hidden units)
#define NTHR 256     // 8 waves of 32
#define BB   64      // batch
#define TT   800     // timesteps
#define HH   400     // hidden
#define KATT 10
#define UU   96
#define CD   64
#define MM   20
#define K1P  480     // 3 + 64 + 400 = 467 -> pad 480
#define K2P  896     // 3 + 400 + 64 + 400 = 867 -> pad 896
#define KMP  1216    // 1200 -> pad 1216

typedef __attribute__((ext_vector_type(16))) __bf16   v16bf;
typedef __attribute__((ext_vector_type(8)))  float    v8f;
typedef __attribute__((ext_vector_type(4)))  unsigned u32x4;

// ---------------- workspace layout (bytes) ----------------
static constexpr size_t SZ_WC1 = (size_t)1600 * K1P * 2;
static constexpr size_t SZ_WC2 = (size_t)1600 * K2P * 2;
static constexpr size_t SZ_WM  = (size_t)128 * KMP * 2;
static constexpr size_t SZ_TE  = (size_t)BB * UU * CD * 4;
static constexpr size_t SZ_A1  = (size_t)2 * BB * K1P * 2;
static constexpr size_t SZ_A2  = (size_t)2 * BB * K2P * 2;
static constexpr size_t SZ_AM  = (size_t)2 * BB * KMP * 2;
static constexpr size_t SZ_H   = (size_t)BB * HH * 4;

static constexpr size_t OFF_WC1 = 0;
static constexpr size_t OFF_WC2 = OFF_WC1 + SZ_WC1;
static constexpr size_t OFF_WC3 = OFF_WC2 + SZ_WC2;
static constexpr size_t OFF_WM  = OFF_WC3 + SZ_WC2;
static constexpr size_t OFF_TE  = OFF_WM  + SZ_WM;
static constexpr size_t OFF_A1  = OFF_TE  + SZ_TE;
static constexpr size_t OFF_A2  = OFF_A1  + SZ_A1;
static constexpr size_t OFF_A3  = OFF_A2  + SZ_A2;
static constexpr size_t OFF_AM  = OFF_A3  + SZ_A2;
static constexpr size_t OFF_H1F = OFF_AM  + SZ_AM;
static constexpr size_t OFF_C1  = OFF_H1F + SZ_H;
static constexpr size_t OFF_C2  = OFF_C1  + SZ_H;
static constexpr size_t OFF_C3  = OFF_C2  + SZ_H;
static constexpr size_t OFF_KAP = OFF_C3  + SZ_H;                 // 64*10*4
static constexpr size_t OFF_BAR = OFF_KAP + (size_t)BB * KATT * 4;

// ---------------- helpers ----------------
__device__ __forceinline__ unsigned short f2bf(float f) {
  unsigned u = __builtin_bit_cast(unsigned, f);
  u += 0x7fffu + ((u >> 16) & 1u);          // round-to-nearest-even
  return (unsigned short)(u >> 16);
}

// Load one 16x32 bf16 fragment (A layout; identical layout serves B when the
// weight matrix is stored row-major N x K). Per ISA 7.12.2:
//   lanes 0-15 : row = lane,    K = [0..7]  and [16..23]
//   lanes 16-31: row = lane-16, K = [8..15] and [24..31]
__device__ __forceinline__ v16bf ldfrag(const unsigned short* p) {
  union FragU { u32x4 q[2]; v16bf v; } f;
  f.q[0] = *(const u32x4*)(p);
  f.q[1] = *(const u32x4*)(p + 16);
  return f.v;
}

__device__ __forceinline__ v8f wmma_bf16(v16bf a, v16bf b, v8f c) {
  return __builtin_amdgcn_wmma_f32_16x16x32_bf16(false, a, false, b,
                                                 (short)0, c, false, false);
}

// monotonic grid barrier (counter zeroed per launch by hipMemsetAsync)
__device__ __forceinline__ void gbar(unsigned* cnt, unsigned target) {
  __threadfence();
  __syncthreads();
  if (threadIdx.x == 0) {
    __hip_atomic_fetch_add(cnt, 1u, __ATOMIC_RELEASE, __HIP_MEMORY_SCOPE_AGENT);
    while (__hip_atomic_load(cnt, __ATOMIC_ACQUIRE, __HIP_MEMORY_SCOPE_AGENT) < target)
      __builtin_amdgcn_s_sleep(1);
  }
  __syncthreads();
}

// gates GEMM: block owns hidden units [h_off, h_off+16); computes 16 tiles
// (4 gates x 4 M-tiles) of the 64 x 1600 gate matrix into LDS gsm[4][64][16].
// Wave wv handles gate g = wv&3 and M-tiles {wv>>2, (wv>>2)+2}: the two tiles
// SHARE the weight (B) fragment -> 1 B-load + 2 A-loads + 2 WMMAs per chunk,
// with two independent accumulator chains.
__device__ __forceinline__ void gemm_gates(const unsigned short* __restrict__ A,
                                           const unsigned short* __restrict__ W,
                                           int strideK, int nchunks, int h_off,
                                           float* gsm) {
  const int lane = threadIdx.x & 31;
  const int wv   = threadIdx.x >> 5;
  const int m16  = lane & 15;
  const int ksel = (lane & 16) ? 8 : 0;
  const int g    = wv & 3;
  const int mt0  = wv >> 2;           // 0 or 1
  const int mt1  = mt0 + 2;           // 2 or 3
  const unsigned short* ap0 = A + (size_t)(mt0 * 16 + m16) * strideK + ksel;
  const unsigned short* ap1 = A + (size_t)(mt1 * 16 + m16) * strideK + ksel;
  const unsigned short* wp  = W + (size_t)(g * HH + h_off + m16) * strideK + ksel;
  v8f acc0 = {};
  v8f acc1 = {};
#pragma clang loop unroll_count(2)
  for (int kc = 0; kc < nchunks; ++kc) {
    v16bf bf = ldfrag(wp  + kc * 32);
    v16bf a0 = ldfrag(ap0 + kc * 32);
    v16bf a1 = ldfrag(ap1 + kc * 32);
    acc0 = wmma_bf16(a0, bf, acc0);
    acc1 = wmma_bf16(a1, bf, acc1);
  }
  const int hsel   = (lane & 16) ? 8 : 0;
  const int n16    = lane & 15;
  const int mbase0 = mt0 * 16 + hsel;
  const int mbase1 = mt1 * 16 + hsel;
#pragma unroll
  for (int r = 0; r < 8; ++r) {
    gsm[g * 1024 + (mbase0 + r) * 16 + n16] = acc0[r];
    gsm[g * 1024 + (mbase1 + r) * 16 + n16] = acc1[r];
  }
}

// ---------------- prep kernel: bf16 weight packing, embeddings, state init ----------------
__global__ void hwsynth_prep(const float* __restrict__ x,
                             const int*   __restrict__ tokens,
                             const float* __restrict__ emb,
                             const float* __restrict__ Wih1, const float* __restrict__ Whh1,
                             const float* __restrict__ Wih2, const float* __restrict__ Whh2,
                             const float* __restrict__ Wih3, const float* __restrict__ Whh3,
                             const float* __restrict__ WmI,
                             unsigned short* wc1, unsigned short* wc2, unsigned short* wc3,
                             unsigned short* wm, float* te,
                             unsigned short* a1, unsigned short* a2, unsigned short* a3,
                             unsigned short* am,
                             float* c1, float* c2, float* c3, float* kap) {
  const size_t gid = (size_t)blockIdx.x * blockDim.x + threadIdx.x;
  const size_t str = (size_t)gridDim.x * blockDim.x;

  // Wc1: K order [x(3), w(64), h1prev(400)] matches W_ih1 input = concat(x,w)
  for (size_t i = gid; i < (size_t)1600 * K1P; i += str) {
    int n = (int)(i / K1P), k = (int)(i % K1P);
    float v = 0.f;
    if (k < 67) v = Wih1[n * 67 + k];
    else if (k < 467) v = Whh1[n * HH + (k - 67)];
    wc1[i] = f2bf(v);
  }
  // Wc2: K order [x(3), h1(400), w(64), h2prev(400)]
  for (size_t i = gid; i < (size_t)1600 * K2P; i += str) {
    int n = (int)(i / K2P), k = (int)(i % K2P);
    float v = 0.f;
    if (k < 467) v = Wih2[n * 467 + k];
    else if (k < 867) v = Whh2[n * HH + (k - 467)];
    wc2[i] = f2bf(v);
  }
  for (size_t i = gid; i < (size_t)1600 * K2P; i += str) {
    int n = (int)(i / K2P), k = (int)(i % K2P);
    float v = 0.f;
    if (k < 467) v = Wih3[n * 467 + k];
    else if (k < 867) v = Whh3[n * HH + (k - 467)];
    wc3[i] = f2bf(v);
  }
  // Wm padded to 128 x 1216
  for (size_t i = gid; i < (size_t)128 * KMP; i += str) {
    int n = (int)(i / KMP), k = (int)(i % KMP);
    wm[i] = f2bf((n < 121 && k < 1200) ? WmI[n * 1200 + k] : 0.f);
  }
  // text_encoded = emb[text_tokens]
  for (size_t i = gid; i < (size_t)BB * UU * CD; i += str) {
    int d = (int)(i % CD);
    int u = (int)((i / CD) % UU);
    int b = (int)(i / ((size_t)CD * UU));
    te[i] = emb[tokens[b * UU + u] * CD + d];
  }
  // A buffers (both parities): zeros; parity 0 gets x_0 in cols 0..2
  for (size_t i = gid; i < (size_t)2 * BB * K1P; i += str) {
    int par = (int)(i / ((size_t)BB * K1P));
    int rr  = (int)(i % ((size_t)BB * K1P));
    int b = rr / K1P, k = rr % K1P;
    float v = (par == 0 && k < 3) ? x[(size_t)b * TT * 3 + k] : 0.f;
    a1[i] = f2bf(v);
  }
  for (size_t i = gid; i < (size_t)2 * BB * K2P; i += str) {
    int par = (int)(i / ((size_t)BB * K2P));
    int rr  = (int)(i % ((size_t)BB * K2P));
    int b = rr / K2P, k = rr % K2P;
    float v = (par == 0 && k < 3) ? x[(size_t)b * TT * 3 + k] : 0.f;
    a2[i] = f2bf(v);
  }
  for (size_t i = gid; i < (size_t)2 * BB * K2P; i += str) {
    int par = (int)(i / ((size_t)BB * K2P));
    int rr  = (int)(i % ((size_t)BB * K2P));
    int b = rr / K2P, k = rr % K2P;
    float v = (par == 0 && k < 3) ? x[(size_t)b * TT * 3 + k] : 0.f;
    a3[i] = f2bf(v);
  }
  for (size_t i = gid; i < (size_t)2 * BB * KMP; i += str) am[i] = 0;
  for (size_t i = gid; i < (size_t)BB * HH; i += str) { c1[i] = 0.f; c2[i] = 0.f; c3[i] = 0.f; }
  for (size_t i = gid; i < (size_t)BB * KATT; i += str) kap[i] = 0.f;
}

// ---------------- persistent recurrent kernel ----------------
__global__ void __launch_bounds__(NTHR)
hwsynth_step(const float* __restrict__ x,
             const float* __restrict__ Ww,  const float* __restrict__ bw,
             const float* __restrict__ bi1, const float* __restrict__ bh1,
             const float* __restrict__ bi2, const float* __restrict__ bh2,
             const float* __restrict__ bi3, const float* __restrict__ bh3,
             const float* __restrict__ bm,
             const unsigned short* __restrict__ wc1,
             const unsigned short* __restrict__ wc2,
             const unsigned short* __restrict__ wc3,
             const unsigned short* __restrict__ wmW,
             const float* __restrict__ te,
             unsigned short* a1, unsigned short* a2, unsigned short* a3, unsigned short* am,
             float* h1f, float* c1, float* c2, float* c3, float* kap,
             unsigned int* bar, float* out) {
  __shared__ float smem[10240];            // 40 KB: gates / attention / res reuse
  const int j = blockIdx.x;                // 0..24
  const int h_off = j * 16;
  const size_t BTn = (size_t)BB * TT;      // 51200
  unsigned bar_iter = 0;

  for (int t = 0; t < TT; ++t) {
    const int cur = t & 1, nxt = cur ^ 1;
    const unsigned short* A1c = a1 + (size_t)cur * BB * K1P;
    unsigned short*       A1n = a1 + (size_t)nxt * BB * K1P;
    unsigned short*       A2c = a2 + (size_t)cur * BB * K2P;
    unsigned short*       A2n = a2 + (size_t)nxt * BB * K2P;
    unsigned short*       A3c = a3 + (size_t)cur * BB * K2P;
    unsigned short*       A3n = a3 + (size_t)nxt * BB * K2P;
    unsigned short*       AMc = am + (size_t)cur * BB * KMP;

    // ---- phase 1: LSTM1 ----
    gemm_gates(A1c, wc1, K1P, K1P / 32, h_off, smem);
    __syncthreads();
#pragma unroll
    for (int r = 0; r < 4; ++r) {
      int idx = (int)threadIdx.x + r * NTHR;     // 0..1023 = (b,u)
      int b = idx >> 4, u = idx & 15;
      int hu = h_off + u;
      float gi = smem[0 * 1024 + b * 16 + u] + bi1[hu]          + bh1[hu];
      float gf = smem[1 * 1024 + b * 16 + u] + bi1[HH + hu]     + bh1[HH + hu];
      float gg = smem[2 * 1024 + b * 16 + u] + bi1[2 * HH + hu] + bh1[2 * HH + hu];
      float go = smem[3 * 1024 + b * 16 + u] + bi1[3 * HH + hu] + bh1[3 * HH + hu];
      float ii = 1.f / (1.f + expf(-gi));
      float ff = 1.f / (1.f + expf(-gf));
      float oo = 1.f / (1.f + expf(-go));
      float c  = ff * c1[b * HH + hu] + ii * tanhf(gg);
      c1[b * HH + hu] = c;
      float h = oo * tanhf(c);
      h1f[b * HH + hu] = h;
      unsigned short hb = f2bf(h);
      A2c[b * K2P + 3 + hu]  = hb;     // LSTM2 input (this step)
      AMc[b * KMP + hu]      = hb;     // mixture-head input (this step)
      A1n[b * K1P + 67 + hu] = hb;     // h1_prev for next step
    }
    if (j == NBLK - 1 && t + 1 < TT && threadIdx.x < BB * 3) {
      int b = threadIdx.x / 3, cc = threadIdx.x % 3;
      unsigned short v = f2bf(x[(size_t)b * TT * 3 + (size_t)(t + 1) * 3 + cc]);
      A1n[b * K1P + cc] = v;
      A2n[b * K2P + cc] = v;
      A3n[b * K2P + cc] = v;
    }
    gbar(bar, (++bar_iter) * NBLK);

    // ---- phase 2: attention window (block 0, VALU fp32) ----
    if (j == 0) {
      float* pS   = smem;           // 64*30
      float* aS   = smem + 2048;    // 64*10
      float* bS   = smem + 2688;
      float* kS   = smem + 3328;
      float* phiS = smem + 4096;    // 64*96
      for (int idx = threadIdx.x; idx < BB * 3 * KATT; idx += NTHR) {
        int b = idx / 30, jj = idx % 30;
        const float* hrow = h1f + (size_t)b * HH;
        const float* wrow = Ww + (size_t)jj * HH;
        float s = bw[jj];
        for (int k = 0; k < HH; ++k) s += hrow[k] * wrow[k];
        pS[b * 30 + jj] = s;
      }
      __syncthreads();
      for (int idx = threadIdx.x; idx < BB * KATT; idx += NTHR) {
        int b = idx / KATT, k = idx % KATT;
        float al = expf(pS[b * 30 + k]);
        float be = expf(pS[b * 30 + 10 + k]);
        float kh = fminf(5.0f, fmaxf(-10.0f, pS[b * 30 + 20 + k]));
        float kp = kap[idx] + expf(kh);
        kap[idx] = kp;
        aS[idx] = al; bS[idx] = be; kS[idx] = kp;
      }
      __syncthreads();
      for (int idx = threadIdx.x; idx < BB * UU; idx += NTHR) {
        int b = idx / UU, u = idx % UU;
        float uf = (float)(u + 1), s = 0.f;
        for (int k = 0; k < KATT; ++k) {
          float d = kS[b * KATT + k] - uf;
          s += aS[b * KATT + k] * expf(-bS[b * KATT + k] * d * d);
        }
        phiS[idx] = s;
      }
      __syncthreads();
      for (int idx = threadIdx.x; idx < BB * CD; idx += NTHR) {
        int b = idx / CD, d = idx % CD;
        const float* tep = te + (size_t)b * UU * CD + d;
        float s = 0.f;
        for (int u = 0; u < UU; ++u) s += phiS[b * UU + u] * tep[u * CD];
        unsigned short v = f2bf(s);
        A1n[b * K1P + 3 + d]   = v;    // w for next step's LSTM1 input
        A2c[b * K2P + 403 + d] = v;    // w for this step's LSTM2
        A3c[b * K2P + 403 + d] = v;    // w for this step's LSTM3
      }
    }
    gbar(bar, (++bar_iter) * NBLK);

    // ---- phase 3: LSTM2 ----
    gemm_gates(A2c, wc2, K2P, K2P / 32, h_off, smem);
    __syncthreads();
#pragma unroll
    for (int r = 0; r < 4; ++r) {
      int idx = (int)threadIdx.x + r * NTHR;
      int b = idx >> 4, u = idx & 15;
      int hu = h_off + u;
      float gi = smem[0 * 1024 + b * 16 + u] + bi2[hu]          + bh2[hu];
      float gf = smem[1 * 1024 + b * 16 + u] + bi2[HH + hu]     + bh2[HH + hu];
      float gg = smem[2 * 1024 + b * 16 + u] + bi2[2 * HH + hu] + bh2[2 * HH + hu];
      float go = smem[3 * 1024 + b * 16 + u] + bi2[3 * HH + hu] + bh2[3 * HH + hu];
      float ii = 1.f / (1.f + expf(-gi));
      float ff = 1.f / (1.f + expf(-gf));
      float oo = 1.f / (1.f + expf(-go));
      float c  = ff * c2[b * HH + hu] + ii * tanhf(gg);
      c2[b * HH + hu] = c;
      float h = oo * tanhf(c);
      unsigned short hb = f2bf(h);
      A3c[b * K2P + 3 + hu]   = hb;    // LSTM3 input (this step)
      AMc[b * KMP + HH + hu]  = hb;    // mixture-head input
      a2[(size_t)nxt * BB * K2P + b * K2P + 467 + hu] = hb;   // h2_prev next step
    }
    gbar(bar, (++bar_iter) * NBLK);

    // ---- phase 4: LSTM3 ----
    gemm_gates(A3c, wc3, K2P, K2P / 32, h_off, smem);
    __syncthreads();
#pragma unroll
    for (int r = 0; r < 4; ++r) {
      int idx = (int)threadIdx.x + r * NTHR;
      int b = idx >> 4, u = idx & 15;
      int hu = h_off + u;
      float gi = smem[0 * 1024 + b * 16 + u] + bi3[hu]          + bh3[hu];
      float gf = smem[1 * 1024 + b * 16 + u] + bi3[HH + hu]     + bh3[HH + hu];
      float gg = smem[2 * 1024 + b * 16 + u] + bi3[2 * HH + hu] + bh3[2 * HH + hu];
      float go = smem[3 * 1024 + b * 16 + u] + bi3[3 * HH + hu] + bh3[3 * HH + hu];
      float ii = 1.f / (1.f + expf(-gi));
      float ff = 1.f / (1.f + expf(-gf));
      float oo = 1.f / (1.f + expf(-go));
      float c  = ff * c3[b * HH + hu] + ii * tanhf(gg);
      c3[b * HH + hu] = c;
      float h = oo * tanhf(c);
      unsigned short hb = f2bf(h);
      AMc[b * KMP + 2 * HH + hu] = hb;
      a3[(size_t)nxt * BB * K2P + b * K2P + 467 + hu] = hb;   // h3_prev next step
    }
    gbar(bar, (++bar_iter) * NBLK);

    // ---- phase 5: mixture head + transforms (blocks 0..3; AM double-buffered) ----
    if (j < 4) {
      __syncthreads();
      const int lane = threadIdx.x & 31, wv = threadIdx.x >> 5;
      const int nt = wv & 1, mt = wv >> 1;
      const int m16 = lane & 15;
      const int ksel = (lane & 16) ? 8 : 0;
      const unsigned short* ap = AMc + (size_t)(mt * 16 + m16) * KMP + ksel;
      const unsigned short* wp = wmW + (size_t)(j * 32 + nt * 16 + m16) * KMP + ksel;
      v8f acc = {};
#pragma clang loop unroll_count(2)
      for (int kc = 0; kc < KMP / 32; ++kc)
        acc = wmma_bf16(ldfrag(ap + kc * 32), ldfrag(wp + kc * 32), acc);
      const int ncol = nt * 16 + (lane & 15);        // 0..31 local column
      const int ngl  = j * 32 + ncol;
      const float bb = (ngl < 121) ? bm[ngl] : 0.f;
      const int mbase = mt * 16 + ((lane & 16) ? 8 : 0);
#pragma unroll
      for (int r = 0; r < 8; ++r)
        smem[(mbase + r) * 32 + ncol] = acc[r] + bb;
      __syncthreads();
      // elementwise transforms for this block's 32 columns
      for (int idx = threadIdx.x; idx < BB * 32; idx += NTHR) {
        int b = idx >> 5, cc = idx & 31;
        int c = j * 32 + cc;
        float v = smem[b * 32 + cc];
        size_t mo = ((size_t)b * TT + t) * MM;
        if (c == 0)        out[(size_t)b * TT + t] = v;                       // eos_hat
        else if (c <= 20)  { /* pi handled by softmax below */ }
        else if (c <= 40)  out[BTn * 21  + mo + (c - 21)]  = v;               // mu_x
        else if (c <= 60)  out[BTn * 41  + mo + (c - 41)]  = v;               // mu_y
        else if (c <= 80)  out[BTn * 61  + mo + (c - 61)]  = expf(v) + 1e-6f; // sigma_x
        else if (c <= 100) out[BTn * 81  + mo + (c - 81)]  = expf(v) + 1e-6f; // sigma_y
        else if (c <= 120) out[BTn * 101 + mo + (c - 101)] = tanhf(v) * 0.999f; // rho
      }
      if (j == 0 && threadIdx.x < BB) {              // pi softmax (cols 1..20)
        int b = threadIdx.x;
        float mx = -1e30f;
        for (int k = 1; k <= MM; ++k) mx = fmaxf(mx, smem[b * 32 + k]);
        float e[MM], s = 0.f;
        for (int k = 0; k < MM; ++k) { e[k] = expf(smem[b * 32 + 1 + k] - mx); s += e[k]; }
        float inv = 1.f / s;
        size_t mo = ((size_t)b * TT + t) * MM;
        for (int k = 0; k < MM; ++k) out[BTn + mo + k] = e[k] * inv;
      }
      __syncthreads();   // protect smem reads before next step's gate stores
    }
  }
}

// ---------------- host launcher ----------------
extern "C" void kernel_launch(void* const* d_in, const int* in_sizes, int n_in,
                              void* d_out, int out_size, void* d_ws, size_t ws_size,
                              hipStream_t stream) {
  (void)in_sizes; (void)n_in; (void)out_size; (void)ws_size;
  const float* x      = (const float*)d_in[0];
  const int*   tokens = (const int*)  d_in[1];
  const float* emb    = (const float*)d_in[2];
  const float* Wih1   = (const float*)d_in[3];
  const float* Whh1   = (const float*)d_in[4];
  const float* bih1   = (const float*)d_in[5];
  const float* bhh1   = (const float*)d_in[6];
  const float* Wih2   = (const float*)d_in[7];
  const float* Whh2   = (const float*)d_in[8];
  const float* bih2   = (const float*)d_in[9];
  const float* bhh2   = (const float*)d_in[10];
  const float* Wih3   = (const float*)d_in[11];
  const float* Whh3   = (const float*)d_in[12];
  const float* bih3   = (const float*)d_in[13];
  const float* bhh3   = (const float*)d_in[14];
  const float* Ww     = (const float*)d_in[15];
  const float* bw     = (const float*)d_in[16];
  const float* Wm     = (const float*)d_in[17];
  const float* bm     = (const float*)d_in[18];

  char* ws = (char*)d_ws;
  unsigned short* wc1 = (unsigned short*)(ws + OFF_WC1);
  unsigned short* wc2 = (unsigned short*)(ws + OFF_WC2);
  unsigned short* wc3 = (unsigned short*)(ws + OFF_WC3);
  unsigned short* wm  = (unsigned short*)(ws + OFF_WM);
  float*          te  = (float*)        (ws + OFF_TE);
  unsigned short* a1  = (unsigned short*)(ws + OFF_A1);
  unsigned short* a2  = (unsigned short*)(ws + OFF_A2);
  unsigned short* a3  = (unsigned short*)(ws + OFF_A3);
  unsigned short* am  = (unsigned short*)(ws + OFF_AM);
  float*          h1f = (float*)(ws + OFF_H1F);
  float*          c1  = (float*)(ws + OFF_C1);
  float*          c2  = (float*)(ws + OFF_C2);
  float*          c3  = (float*)(ws + OFF_C3);
  float*          kap = (float*)(ws + OFF_KAP);
  unsigned int*   bar = (unsigned int*)(ws + OFF_BAR);

  hipMemsetAsync(bar, 0, 256, stream);   // deterministic barrier reset each call
  hwsynth_prep<<<512, NTHR, 0, stream>>>(x, tokens, emb, Wih1, Whh1, Wih2, Whh2,
                                         Wih3, Whh3, Wm, wc1, wc2, wc3, wm, te,
                                         a1, a2, a3, am, c1, c2, c3, kap);
  hwsynth_step<<<NBLK, NTHR, 0, stream>>>(x, Ww, bw, bih1, bhh1, bih2, bhh2,
                                          bih3, bhh3, bm, wc1, wc2, wc3, wm, te,
                                          a1, a2, a3, am, h1f, c1, c2, c3, kap,
                                          bar, (float*)d_out);
}
